// AttentionDecoder_11519102288628
// MI455X (gfx1250) — matrix operand order repaired and checked by reading, not compiled, not run on hardware
//
#include <hip/hip_runtime.h>
#include <hip/hip_bf16.h>
#include <math.h>

// ---------------------------------------------------------------------------
// Dimensions
// ---------------------------------------------------------------------------
constexpr int BB   = 32;    // batch
constexpr int TDEC = 200;   // decoder steps
constexpr int TENC = 256;   // encoder steps
constexpr int ENC  = 256;   // encoder dim
constexpr int HD   = 256;   // GRU hidden
constexpr int HD3  = 768;   // 3*GRU
constexpr int DIN  = 400;   // n_mels * r
constexpr int CIN  = 656;   // DIN + HD
constexpr int CINP = 672;   // CIN padded to mult of 32
constexpr int NMR  = 400;   // output dim

// ---------------------------------------------------------------------------
// WMMA bf16 helpers (gfx1250: V_WMMA_F32_16X16X32_BF16, wave32)
// ---------------------------------------------------------------------------
typedef __attribute__((ext_vector_type(16))) __bf16 bf16x16;
typedef __attribute__((ext_vector_type(8)))  float  f32x8;

__device__ __forceinline__ f32x8 wmma_bf16(bf16x16 a, bf16x16 b, f32x8 c) {
  // 8 args: (neg_a, A, neg_b, B, c_mod, C, reuse_a, reuse_b)
  return __builtin_amdgcn_wmma_f32_16x16x32_bf16(false, a, false, b, (short)0, c,
                                                 false, false);
}

union Frag32B { bf16x16 v; uint4 q[2]; };

// A fragment 16x32 bf16 from LDS (row-major, ldE elements/row, 16B-aligned rows)
// lane L: m = L&15, kb = (L>>4)*8 ; dw 0..3 -> k = kb..kb+7 ; dw 4..7 -> k = 16+kb..
__device__ __forceinline__ bf16x16 load_a_frag(const __bf16* tile, int ldE, int kOff) {
  int lane = threadIdx.x & 31;
  int m  = lane & 15;
  int kb = (lane >> 4) * 8;
  const __bf16* p = tile + m * ldE + kOff + kb;
  Frag32B u;
  u.q[0] = *(const uint4*)(p);        // k = kb .. kb+7
  u.q[1] = *(const uint4*)(p + 16);   // k = 16+kb .. 16+kb+7
  return u.v;
}

// B fragment 32x16 bf16 from pre-packed (fragment-major) global memory:
// each lane reads its 16 bf16 contiguously (2 x b128).
__device__ __forceinline__ bf16x16 load_b_frag(const __bf16* packedTile) {
  int lane = threadIdx.x & 31;
  const __bf16* p = packedTile + lane * 16;
  Frag32B u;
  u.q[0] = *(const uint4*)(p);
  u.q[1] = *(const uint4*)(p + 8);
  return u.v;
}

// One 16x16 output tile, accumulating over nkt K-tiles (K = 32*nkt)
__device__ __forceinline__ f32x8 gemm_tile(const __bf16* sA, int ldE,
                                           const __bf16* Bp, int nt, int nkt) {
  f32x8 acc = {0.f, 0.f, 0.f, 0.f, 0.f, 0.f, 0.f, 0.f};
  const __bf16* bt = Bp + (size_t)nt * nkt * 512;
  for (int kt = 0; kt < nkt; ++kt) {
    bf16x16 a = load_a_frag(sA, ldE, kt * 32);
    bf16x16 b = load_b_frag(bt + (size_t)kt * 512);
    acc = wmma_bf16(a, b, acc);
  }
  return acc;
}

__device__ __forceinline__ float sigmoidf_(float x) {
  return 1.f / (1.f + __expf(-x));
}

// ---------------------------------------------------------------------------
// Workspace layout (bytes)
// ---------------------------------------------------------------------------
constexpr size_t OFF_WXD  = 0;                         // 21*48*512*2 = 1032192
constexpr size_t OFF_WHD  = OFF_WXD + 1032192;         //  8*48*512*2 =  393216
constexpr size_t OFF_WQ   = OFF_WHD + 393216;          //  8*16*512*2 =  131072
constexpr size_t OFF_WAL  = OFF_WQ  + 131072;          // 16*16*512*2 =  262144
constexpr size_t OFF_WX1  = OFF_WAL + 262144;
constexpr size_t OFF_WH1  = OFF_WX1 + 393216;
constexpr size_t OFF_WX2  = OFF_WH1 + 393216;
constexpr size_t OFF_WH2  = OFF_WX2 + 393216;
constexpr size_t OFF_WOUT = OFF_WH2 + 393216;          //  8*25*512*2 =  204800
constexpr size_t OFF_WM   = OFF_WOUT + 204800;         //  8*16*512*2 =  131072
constexpr size_t OFF_MEMBF= OFF_WM  + 131072;          // 32*256*256*2 = 4194304
constexpr size_t OFF_KEYS = OFF_MEMBF + 4194304;       // 32*256*256*4 = 8388608
constexpr size_t OFF_ATTN = OFF_KEYS + 8388608;        // 200*32*256*4 = 6553600
constexpr size_t OFF_OUT1 = OFF_ATTN + 6553600;
constexpr size_t OFF_OUT2 = OFF_OUT1 + 6553600;

// ---------------------------------------------------------------------------
// Pack fp32 weight [K x N] -> bf16 fragment-major tiles (nt-major, kt-minor).
// Within a 32x16 tile, element order = lane*16 + e, with n = lane&15,
// k = (lane>>4)*8 + e for e<8, 16 + (lane>>4)*8 + (e-8) for e>=8  (matches A-fragment k order).
// ---------------------------------------------------------------------------
__global__ void k_pack_bf16(const float* __restrict__ src, __bf16* __restrict__ dst,
                            int K, int N, int Kpad) {
  int nkt = Kpad >> 5;
  long total = (long)nkt * (N >> 4) * 512;
  for (long p = (long)blockIdx.x * blockDim.x + threadIdx.x; p < total;
       p += (long)gridDim.x * blockDim.x) {
    int  w    = (int)(p & 511);
    long tile = p >> 9;
    int  kt   = (int)(tile % nkt);
    int  nt   = (int)(tile / nkt);
    int  lane = w >> 4, e = w & 15;
    int  n = nt * 16 + (lane & 15);
    int  k = kt * 32 + (e < 8 ? (lane >> 4) * 8 + e : 16 + (lane >> 4) * 8 + (e - 8));
    float v = (k < K) ? src[(size_t)k * N + n] : 0.f;
    dst[p] = (__bf16)v;
  }
}

__global__ void k_cvt_bf16(const float* __restrict__ src, __bf16* __restrict__ dst,
                           long n) {
  for (long i = (long)blockIdx.x * blockDim.x + threadIdx.x; i < n;
       i += (long)gridDim.x * blockDim.x)
    dst[i] = (__bf16)src[i];
}

// ---------------------------------------------------------------------------
// keys = memory @ Wm : [8192 x 256] x [256 x 256] -> fp32
// block = 256 thr (8 waves), one 16-row M-tile / block, wave -> 2 N-tiles
// ---------------------------------------------------------------------------
__global__ void k_keys(const float* __restrict__ mem, const __bf16* __restrict__ WmP,
                       float* __restrict__ keys) {
  extern __shared__ char smem[];
  __bf16* sA = (__bf16*)smem;  // 16 x 256 bf16
  int tid = threadIdx.x, wv = tid >> 5, lane = tid & 31;
  int m0 = blockIdx.x * 16;
  for (int i = tid; i < 16 * ENC; i += 256)
    sA[i] = (__bf16)mem[(size_t)(m0 + (i >> 8)) * ENC + (i & 255)];
  __syncthreads();
  int n = lane & 15, mb = (lane >> 4) * 8;
  for (int j = 0; j < 2; ++j) {
    int nt = wv * 2 + j;
    f32x8 acc = gemm_tile(sA, ENC, WmP, nt, 8);
#pragma unroll
    for (int v = 0; v < 8; ++v)
      keys[(size_t)(m0 + mb + v) * HD + nt * 16 + n] = acc[v];
  }
}

// ---------------------------------------------------------------------------
// Persistent attention-decoder scan. grid = 2 blocks (16 batch rows each),
// block = 256 threads = 8 wave32s. All recurrence state lives in LDS.
// ---------------------------------------------------------------------------
constexpr int S_A   = 0;                  // 16*672 bf16 = 21504 (A staging, reused)
constexpr int S_G   = S_A   + 21504;      // 16*768 f32  = 49152 (gx + gh_{z,r})
constexpr int S_GHN = S_G   + 49152;      // 16*256 f32  = 16384 (gh_n)
constexpr int S_H   = S_GHN + 16384;      // 16*256 f32
constexpr int S_HBF = S_H   + 16384;      // 16*256 bf16 = 8192
constexpr int S_ABF = S_HBF + 8192;       // 16*256 bf16 = 8192 (prev attention)
constexpr int S_Q   = S_ABF + 8192;       // 16*256 f32
constexpr int S_SC  = S_Q   + 16384;      // 16*256 f32 (scores -> align)
constexpr int S_CTX = S_SC  + 16384;      // 16*256 f32
constexpr int S_VA  = S_CTX + 16384;      // 256 f32 = 1024
constexpr int S_MX  = S_VA  + 1024;       // 16 f32 (pad 64)
constexpr int S_INV = S_MX  + 64;
constexpr int SMEM_SCAN = S_INV + 64;     // = 170112 bytes (fits 320KB WGP LDS)

__global__ void k_scan(const float* __restrict__ inputs,
                       const float* __restrict__ keys,
                       const __bf16* __restrict__ memBf,
                       const float* __restrict__ v_a,
                       const float* __restrict__ b_dec,
                       const __bf16* __restrict__ WxdP,
                       const __bf16* __restrict__ WhdP,
                       const __bf16* __restrict__ WqP,
                       const __bf16* __restrict__ WalP,
                       float* __restrict__ attnSeq,
                       float* __restrict__ outAlign) {
  extern __shared__ char smem[];
  __bf16* sA   = (__bf16*)(smem + S_A);
  float*  sG   = (float*)(smem + S_G);
  float*  sGhn = (float*)(smem + S_GHN);
  float*  sH   = (float*)(smem + S_H);
  __bf16* sHbf = (__bf16*)(smem + S_HBF);
  __bf16* sAbf = (__bf16*)(smem + S_ABF);
  float*  sQ   = (float*)(smem + S_Q);
  float*  sSc  = (float*)(smem + S_SC);
  float*  sCtx = (float*)(smem + S_CTX);
  float*  sVa  = (float*)(smem + S_VA);
  float*  sMx  = (float*)(smem + S_MX);
  float*  sInv = (float*)(smem + S_INV);

  int tid = threadIdx.x, wv = tid >> 5, lane = tid & 31;
  int b0 = blockIdx.x * 16;
  int n = lane & 15, mb = (lane >> 4) * 8;

  for (int i = tid; i < 16 * HD; i += 256) {
    sH[i] = 0.f; sHbf[i] = (__bf16)0.f; sAbf[i] = (__bf16)0.f;
  }
  for (int i = tid; i < HD; i += 256) sVa[i] = v_a[i];
  __syncthreads();

  for (int t = 0; t < TDEC; ++t) {
    // (0) build cin = [x_t | prev_attn] as bf16 A tile (16 x 672, zero-padded)
    for (int i = tid; i < 16 * CINP; i += 256) {
      int r = i / CINP, c = i - r * CINP;
      __bf16 v;
      if (c < DIN)       v = (__bf16)inputs[((size_t)(b0 + r) * TDEC + t) * DIN + c];
      else if (c < CIN)  v = sAbf[r * HD + (c - DIN)];
      else               v = (__bf16)0.f;
      sA[i] = v;
    }
    __syncthreads();

    // (1) gx = cin @ Wx_dec : 48 N-tiles, 21 K-tiles, wave -> 6 tiles
    for (int i = 0; i < 6; ++i) {
      int nt = wv * 6 + i;
      f32x8 acc = gemm_tile(sA, CINP, WxdP, nt, 21);
#pragma unroll
      for (int v = 0; v < 8; ++v) sG[(mb + v) * HD3 + nt * 16 + n] = acc[v];
    }
    __syncthreads();

    // (2) gh = h @ Wh_dec : z,r accumulate into sG, n -> sGhn
    for (int i = 0; i < 6; ++i) {
      int nt = wv * 6 + i;
      f32x8 acc = gemm_tile(sHbf, HD, WhdP, nt, 8);
      if (nt < 32) {
#pragma unroll
        for (int v = 0; v < 8; ++v) sG[(mb + v) * HD3 + nt * 16 + n] += acc[v];
      } else {
#pragma unroll
        for (int v = 0; v < 8; ++v) sGhn[(mb + v) * HD + (nt - 32) * 16 + n] = acc[v];
      }
    }
    __syncthreads();

    // (3) GRU gate math -> new h
    for (int i = tid; i < 16 * HD; i += 256) {
      int r = i >> 8, j = i & 255;
      float z  = sigmoidf_(sG[r * HD3 + j] + b_dec[j]);
      float rr = sigmoidf_(sG[r * HD3 + HD + j] + b_dec[HD + j]);
      float nn = tanhf(sG[r * HD3 + 2 * HD + j] + b_dec[2 * HD + j] + rr * sGhn[i]);
      float hn = z * sH[i] + (1.f - z) * nn;
      sH[i] = hn; sHbf[i] = (__bf16)hn;
    }
    __syncthreads();

    // (4) q = h @ Wq : 16 N-tiles, wave -> 2
    for (int j = 0; j < 2; ++j) {
      int nt = wv * 2 + j;
      f32x8 acc = gemm_tile(sHbf, HD, WqP, nt, 8);
#pragma unroll
      for (int v = 0; v < 8; ++v) sQ[(mb + v) * HD + nt * 16 + n] = acc[v];
    }
    __syncthreads();

    // (5) scores[r,e] = sum_a tanh(keys[b,e,a] + q[r,a]) * v_a[a]
    {
      int e = tid;  // 256 threads <-> 256 encoder positions
      for (int r = 0; r < 16; ++r) {
        const float* kr = keys + ((size_t)(b0 + r) * TENC + e) * HD;
        const float* qr = sQ + r * HD;
        float acc = 0.f;
        for (int a = 0; a < HD; ++a) acc += tanhf(kr[a] + qr[a]) * sVa[a];
        sSc[r * TENC + e] = acc;
      }
    }
    __syncthreads();

    // (6) softmax over encoder dim + emit alignments
    if (tid < 16) {
      float mx = -1e30f;
      for (int e = 0; e < TENC; ++e) mx = fmaxf(mx, sSc[tid * TENC + e]);
      float s = 0.f;
      for (int e = 0; e < TENC; ++e) s += __expf(sSc[tid * TENC + e] - mx);
      sMx[tid] = mx; sInv[tid] = 1.f / s;
    }
    __syncthreads();
    for (int i = tid; i < 16 * TENC; i += 256) {
      int r = i >> 8, e = i & 255;
      float v = __expf(sSc[i] - sMx[r]) * sInv[r];
      sSc[i] = v;
      outAlign[((size_t)(b0 + r) * TENC + e) * TDEC + t] = v;
    }
    __syncthreads();

    // (7) ctx[r,d] = sum_e align[r,e] * memory[b,e,d]  (per-batch B: VALU)
    {
      int d = tid;
      for (int r = 0; r < 16; ++r) {
        const __bf16* mrow = memBf + (size_t)(b0 + r) * TENC * ENC + d;
        const float* ar = sSc + r * TENC;
        float acc = 0.f;
        for (int e = 0; e < TENC; ++e) acc += ar[e] * (float)mrow[(size_t)e * ENC];
        sCtx[r * HD + d] = acc;
      }
    }
    __syncthreads();

    // (8) attn = [h | ctx] @ W_al  (16 x 512 A tile, 16 K-tiles, 16 N-tiles)
    for (int i = tid; i < 16 * 512; i += 256) {
      int r = i >> 9, c = i & 511;
      sA[r * 512 + c] = (c < HD) ? sHbf[r * HD + c] : (__bf16)sCtx[r * HD + (c - HD)];
    }
    __syncthreads();
    for (int j = 0; j < 2; ++j) {
      int nt = wv * 2 + j;
      f32x8 acc = gemm_tile(sA, 512, WalP, nt, 16);
#pragma unroll
      for (int v = 0; v < 8; ++v) {
        float val = acc[v];
        attnSeq[((size_t)t * BB + b0 + mb + v) * HD + nt * 16 + n] = val;
        sAbf[(mb + v) * HD + nt * 16 + n] = (__bf16)val;
      }
    }
    __syncthreads();
  }
}

// ---------------------------------------------------------------------------
// Persistent residual GRU: y[t] = x[t] + GRU(x)[t].  grid = 2 blocks.
// ---------------------------------------------------------------------------
constexpr int GS_AX  = 0;                 // 16*256 bf16 = 8192
constexpr int GS_HBF = GS_AX + 8192;      // 16*256 bf16 = 8192
constexpr int GS_G   = GS_HBF + 8192;     // 16*768 f32  = 49152
constexpr int GS_GHN = GS_G + 49152;      // 16*256 f32  = 16384
constexpr int GS_H   = GS_GHN + 16384;    // 16*256 f32  = 16384
constexpr int SMEM_GRU = GS_H + 16384;    // 98304

__global__ void k_gru(const float* __restrict__ xseq, float* __restrict__ yseq,
                      const __bf16* __restrict__ WxP, const __bf16* __restrict__ WhP,
                      const float* __restrict__ bias) {
  extern __shared__ char smem[];
  __bf16* sAx  = (__bf16*)(smem + GS_AX);
  __bf16* sHbf = (__bf16*)(smem + GS_HBF);
  float*  sG   = (float*)(smem + GS_G);
  float*  sGhn = (float*)(smem + GS_GHN);
  float*  sH   = (float*)(smem + GS_H);
  int tid = threadIdx.x, wv = tid >> 5, lane = tid & 31;
  int b0 = blockIdx.x * 16;
  int n = lane & 15, mb = (lane >> 4) * 8;

  for (int i = tid; i < 16 * HD; i += 256) { sHbf[i] = (__bf16)0.f; sH[i] = 0.f; }
  __syncthreads();

  for (int t = 0; t < TDEC; ++t) {
    const float* xt = xseq + ((size_t)t * BB + b0) * HD;
    for (int i = tid; i < 16 * HD; i += 256) sAx[i] = (__bf16)xt[i];
    __syncthreads();
    for (int i = 0; i < 6; ++i) {
      int nt = wv * 6 + i;
      f32x8 acc = gemm_tile(sAx, HD, WxP, nt, 8);
#pragma unroll
      for (int v = 0; v < 8; ++v) sG[(mb + v) * HD3 + nt * 16 + n] = acc[v];
    }
    __syncthreads();
    for (int i = 0; i < 6; ++i) {
      int nt = wv * 6 + i;
      f32x8 acc = gemm_tile(sHbf, HD, WhP, nt, 8);
      if (nt < 32) {
#pragma unroll
        for (int v = 0; v < 8; ++v) sG[(mb + v) * HD3 + nt * 16 + n] += acc[v];
      } else {
#pragma unroll
        for (int v = 0; v < 8; ++v) sGhn[(mb + v) * HD + (nt - 32) * 16 + n] = acc[v];
      }
    }
    __syncthreads();
    for (int i = tid; i < 16 * HD; i += 256) {
      int r = i >> 8, j = i & 255;
      float z  = sigmoidf_(sG[r * HD3 + j] + bias[j]);
      float rr = sigmoidf_(sG[r * HD3 + HD + j] + bias[HD + j]);
      float nn = tanhf(sG[r * HD3 + 2 * HD + j] + bias[2 * HD + j] + rr * sGhn[i]);
      float hn = z * sH[i] + (1.f - z) * nn;
      sH[i] = hn; sHbf[i] = (__bf16)hn;
      yseq[((size_t)t * BB + b0 + r) * HD + j] = xt[i] + hn;
    }
    __syncthreads();
  }
}

// ---------------------------------------------------------------------------
// mel = out2 @ W_out + b_out, stored transposed [B, T, 400].
// grid = 400 blocks (M-tiles over T*B=6400 rows); 25 N-tiles.
// ---------------------------------------------------------------------------
__global__ void k_dense(const float* __restrict__ xseq, const __bf16* __restrict__ WoP,
                        const float* __restrict__ b_out, float* __restrict__ mel) {
  extern __shared__ char smem[];
  __bf16* sA = (__bf16*)smem;  // 16 x 256 bf16
  int tid = threadIdx.x, wv = tid >> 5, lane = tid & 31;
  int m0 = blockIdx.x * 16;
  for (int i = tid; i < 16 * HD; i += 256)
    sA[i] = (__bf16)xseq[(size_t)(m0 + (i >> 8)) * HD + (i & 255)];
  __syncthreads();
  int n = lane & 15, mb = (lane >> 4) * 8;
  for (int j = 0; j < 4; ++j) {
    int nt = wv + j * 8;
    if (nt >= 25) break;
    f32x8 acc = gemm_tile(sA, HD, WoP, nt, 8);
#pragma unroll
    for (int v = 0; v < 8; ++v) {
      int g = m0 + mb + v;
      int tt = g >> 5, b = g & 31;
      mel[((size_t)b * TDEC + tt) * NMR + nt * 16 + n] = acc[v] + b_out[nt * 16 + n];
    }
  }
}

// ---------------------------------------------------------------------------
// Launch
// ---------------------------------------------------------------------------
extern "C" void kernel_launch(void* const* d_in, const int* in_sizes, int n_in,
                              void* d_out, int out_size, void* d_ws, size_t ws_size,
                              hipStream_t stream) {
  (void)in_sizes; (void)n_in; (void)out_size; (void)ws_size;
  const float* inputs = (const float*)d_in[0];
  const float* memory = (const float*)d_in[1];
  const float* Wq     = (const float*)d_in[2];
  const float* Wm     = (const float*)d_in[3];
  const float* v_a    = (const float*)d_in[4];
  const float* W_al   = (const float*)d_in[5];
  const float* Wx_dec = (const float*)d_in[6];
  const float* Wh_dec = (const float*)d_in[7];
  const float* b_dec  = (const float*)d_in[8];
  const float* Wx1    = (const float*)d_in[9];
  const float* Wh1    = (const float*)d_in[10];
  const float* b1     = (const float*)d_in[11];
  const float* Wx2    = (const float*)d_in[12];
  const float* Wh2    = (const float*)d_in[13];
  const float* b2     = (const float*)d_in[14];
  const float* W_out  = (const float*)d_in[15];
  const float* b_out  = (const float*)d_in[16];

  char* ws = (char*)d_ws;
  __bf16* WXD  = (__bf16*)(ws + OFF_WXD);
  __bf16* WHD  = (__bf16*)(ws + OFF_WHD);
  __bf16* WQP  = (__bf16*)(ws + OFF_WQ);
  __bf16* WAL  = (__bf16*)(ws + OFF_WAL);
  __bf16* WX1P = (__bf16*)(ws + OFF_WX1);
  __bf16* WH1P = (__bf16*)(ws + OFF_WH1);
  __bf16* WX2P = (__bf16*)(ws + OFF_WX2);
  __bf16* WH2P = (__bf16*)(ws + OFF_WH2);
  __bf16* WOUT = (__bf16*)(ws + OFF_WOUT);
  __bf16* WMP  = (__bf16*)(ws + OFF_WM);
  __bf16* MEMB = (__bf16*)(ws + OFF_MEMBF);
  float*  KEYS = (float*)(ws + OFF_KEYS);
  float*  ATTN = (float*)(ws + OFF_ATTN);
  float*  OUT1 = (float*)(ws + OFF_OUT1);
  float*  OUT2 = (float*)(ws + OFF_OUT2);

  float* out_mel   = (float*)d_out;
  float* out_align = out_mel + (size_t)BB * TDEC * NMR;

  auto pack = [&](const float* s, __bf16* d, int K, int N, int Kpad) {
    long total = (long)(Kpad / 32) * (N / 16) * 512;
    int blocks = (int)((total + 255) / 256);
    if (blocks > 2048) blocks = 2048;
    k_pack_bf16<<<blocks, 256, 0, stream>>>(s, d, K, N, Kpad);
  };
  pack(Wx_dec, WXD, CIN, HD3, CINP);
  pack(Wh_dec, WHD, HD, HD3, HD);
  pack(Wq,     WQP, HD, HD,  HD);
  pack(W_al,   WAL, 512, HD, 512);
  pack(Wx1,    WX1P, HD, HD3, HD);
  pack(Wh1,    WH1P, HD, HD3, HD);
  pack(Wx2,    WX2P, HD, HD3, HD);
  pack(Wh2,    WH2P, HD, HD3, HD);
  pack(W_out,  WOUT, HD, NMR, HD);
  pack(Wm,     WMP,  ENC, HD, ENC);

  long memN = (long)BB * TENC * ENC;
  k_cvt_bf16<<<2048, 256, 0, stream>>>(memory, MEMB, memN);

  // keys = memory @ Wm   (parallel WMMA GEMM)
  k_keys<<<(BB * TENC) / 16, 256, 16 * ENC * 2, stream>>>(memory, WMP, KEYS);

  // attention decoder scan (persistent, 2 blocks / batch halves)
  k_scan<<<2, 256, SMEM_SCAN, stream>>>(inputs, KEYS, MEMB, v_a, b_dec,
                                        WXD, WHD, WQP, WAL, ATTN, out_align);

  // residual GRUs
  k_gru<<<2, 256, SMEM_GRU, stream>>>(ATTN, OUT1, WX1P, WH1P, b1);
  k_gru<<<2, 256, SMEM_GRU, stream>>>(OUT1, OUT2, WX2P, WH2P, b2);

  // output dense + transpose
  k_dense<<<(TDEC * BB) / 16, 256, 16 * HD * 2, stream>>>(OUT2, WOUT, b_out, out_mel);
}
